// TinyTransformerDecoder_60241211294395
// MI455X (gfx1250) — compile-verified, matrix operand order checked by reading, not run on hardware
//
#include <hip/hip_runtime.h>
#include <hip/hip_bf16.h>

// ---------------------------------------------------------------------------
// Types for CDNA5 WMMA (wave32, 16x16x32 f16 -> f32)
// ---------------------------------------------------------------------------
typedef __attribute__((ext_vector_type(16))) _Float16 v16h;
typedef __attribute__((ext_vector_type(8)))  _Float16 h8;
typedef __attribute__((ext_vector_type(8)))  float    v8f;
typedef __attribute__((ext_vector_type(4)))  float    f4;

#define TM 128
#define TN 64
#define TK 32

// Problem constants (from the reference)
#define BATCH 2
#define SEQ   2048
#define DMODEL 1024
#define NHEAD 16
#define HDIM  64
#define FFDIM 2048
#define VOCAB 32000

// fragment: two contiguous 8xf16 runs per lane (ISA 7.12.2 16-bit layout)
__device__ __forceinline__ v16h load_frag(const _Float16* rowp, int khalf)
{
    h8 lo = *(const h8*)(rowp + khalf * 8);
    h8 hi = *(const h8*)(rowp + 16 + khalf * 8);
    return __builtin_shufflevector(lo, hi, 0,1,2,3,4,5,6,7,8,9,10,11,12,13,14,15);
}

// ---------------------------------------------------------------------------
// Batched GEMM: C = alpha * A(MxK) * B(KxN) [+ bias] [ReLU]
// REQUIRES: M % 128 == 0, N % 64 == 0, K % 32 == 0 (true for every call here)
// 256 threads = 8 wave32s; WG tile 128(M) x 64(N); each wave owns a 32x32
// output block = four 16x16 f32 WMMA accumulators; K stepped by 32.
// Double-buffered LDS staging (f32 -> f16; B pre-transposed to [n][k]):
// one barrier per K-step, next tile's global loads overlap current WMMAs.
// ---------------------------------------------------------------------------
template <int TRANSB>
__global__ __launch_bounds__(256)
void gemm_f16wmma(const float* __restrict__ A, const float* __restrict__ B,
                  float* __restrict__ C, const float* __restrict__ bias,
                  int M, int N, int K, int lda, int ldb, int ldc,
                  float alpha, int relu,
                  long long strideA, long long strideB, long long strideC)
{
    __shared__ __attribute__((aligned(16))) _Float16 As[2][TM][TK + 8];
    __shared__ __attribute__((aligned(16))) _Float16 Bs[2][TN][TK + 8];

    const int bx = blockIdx.x, by = blockIdx.y, bz = blockIdx.z;
    const float* Ab = A + (long long)bz * strideA;
    const float* Bb = B + (long long)bz * strideB;
    float*       Cb = C + (long long)bz * strideC;

    const int tid   = threadIdx.x;
    const int lane  = tid & 31;
    const int wave  = tid >> 5;
    const int mw    = wave & 3;   // 0..3 : which 32-row block
    const int nw    = wave >> 2;  // 0..1 : which 32-col block
    const int row16 = lane & 15;
    const int khalf = lane >> 4;

    const int rowBase = by * TM;
    const int colBase = bx * TN;

    // A staging map: thread t -> row t/2, cols (t&1)*16 .. +16 (4x float4)
    const int ar = tid >> 1;
    const int ac = (tid & 1) * 16;
    // B staging maps (8 floats / thread = 2x float4)
    const int bnT = tid >> 2;            // TRANSB:  row n  (0..63)
    const int bkT = (tid & 3) * 8;       //          col k
    const int bkN = tid >> 3;            // !TRANSB: row k  (0..31)
    const int bnN = (tid & 7) * 8;       //          col n

    const float* aptr = Ab + (long long)(rowBase + ar) * lda + ac;
    const float* bptrT = Bb + (long long)(colBase + bnT) * ldb + bkT;
    const float* bptrN = Bb + (long long)bkN * ldb + colBase + bnN;

    auto load_tile = [&](int k0, f4* ra, f4* rb) {
        const float* ap = aptr + k0;
        ra[0] = *(const f4*)(ap + 0);
        ra[1] = *(const f4*)(ap + 4);
        ra[2] = *(const f4*)(ap + 8);
        ra[3] = *(const f4*)(ap + 12);
        if (TRANSB) {
            const float* bp = bptrT + k0;
            rb[0] = *(const f4*)(bp + 0);
            rb[1] = *(const f4*)(bp + 4);
        } else {
            const float* bp = bptrN + (long long)k0 * ldb;
            rb[0] = *(const f4*)(bp + 0);
            rb[1] = *(const f4*)(bp + 4);
        }
    };

    auto store_tile = [&](int buf, const f4* ra, const f4* rb) {
        h8 ha0, ha1;
        #pragma unroll
        for (int j = 0; j < 4; ++j) {
            ha0[j]     = (_Float16)ra[0][j];
            ha0[4 + j] = (_Float16)ra[1][j];
            ha1[j]     = (_Float16)ra[2][j];
            ha1[4 + j] = (_Float16)ra[3][j];
        }
        *(h8*)&As[buf][ar][ac]     = ha0;
        *(h8*)&As[buf][ar][ac + 8] = ha1;
        if (TRANSB) {
            h8 hb;
            #pragma unroll
            for (int j = 0; j < 4; ++j) {
                hb[j]     = (_Float16)rb[0][j];
                hb[4 + j] = (_Float16)rb[1][j];
            }
            *(h8*)&Bs[buf][bnT][bkT] = hb;
        } else {
            #pragma unroll
            for (int j = 0; j < 4; ++j) Bs[buf][bnN + j][bkN]     = (_Float16)rb[0][j];
            #pragma unroll
            for (int j = 0; j < 4; ++j) Bs[buf][bnN + 4 + j][bkN] = (_Float16)rb[1][j];
        }
    };

    v8f acc00 = {}, acc01 = {}, acc10 = {}, acc11 = {};

    // prologue: stage tile 0 into buffer 0
    {
        f4 ra[4], rb[2];
        load_tile(0, ra, rb);
        store_tile(0, ra, rb);
    }
    __syncthreads();

    const int nsteps = K / TK;
    const int arow0 = mw * 32 + row16;
    const int bcol0 = nw * 32 + row16;

    for (int step = 0; step < nsteps; ++step) {
        const int cur = step & 1;
        const bool hasNext = (step + 1) < nsteps;

        // issue next tile's global loads before this tile's math
        f4 na[4], nb[2];
        if (hasNext) {
            load_tile((step + 1) * TK, na, nb);
            if (step + 2 < nsteps)
                __builtin_prefetch(aptr + (step + 2) * TK, 0, 0);
        }

        // fragments: 8x ds_load_b128, 4x v_wmma per K-step
        v16h af0 = load_frag(As[cur][arow0],      khalf);
        v16h af1 = load_frag(As[cur][arow0 + 16], khalf);
        v16h bf0 = load_frag(Bs[cur][bcol0],      khalf);
        v16h bf1 = load_frag(Bs[cur][bcol0 + 16], khalf);

        acc00 = __builtin_amdgcn_wmma_f32_16x16x32_f16(false, af0, false, bf0, (short)0, acc00, false, false);
        acc01 = __builtin_amdgcn_wmma_f32_16x16x32_f16(false, af0, false, bf1, (short)0, acc01, false, false);
        acc10 = __builtin_amdgcn_wmma_f32_16x16x32_f16(false, af1, false, bf0, (short)0, acc10, false, false);
        acc11 = __builtin_amdgcn_wmma_f32_16x16x32_f16(false, af1, false, bf1, (short)0, acc11, false, false);

        if (hasNext) {
            store_tile(cur ^ 1, na, nb);   // other buffer: no readers this step
            __syncthreads();               // publish before next step's reads
        }
    }

    // ---- epilogue: C/D layout — lane 0-15: N=lane, M=r ; lane 16-31: M=8+r
    const int rB = rowBase + mw * 32 + khalf * 8;
    const int cB = colBase + nw * 32 + row16;
    {
        float bv0 = bias ? bias[cB]      : 0.0f;
        float bv1 = bias ? bias[cB + 16] : 0.0f;
        #pragma unroll
        for (int r = 0; r < 8; ++r) {
            float v00 = acc00[r] * alpha + bv0;
            float v01 = acc01[r] * alpha + bv1;
            float v10 = acc10[r] * alpha + bv0;
            float v11 = acc11[r] * alpha + bv1;
            if (relu) {
                v00 = fmaxf(v00, 0.0f); v01 = fmaxf(v01, 0.0f);
                v10 = fmaxf(v10, 0.0f); v11 = fmaxf(v11, 0.0f);
            }
            Cb[(long long)(rB + r)      * ldc + cB]      = v00;
            Cb[(long long)(rB + r)      * ldc + cB + 16] = v01;
            Cb[(long long)(rB + r + 16) * ldc + cB]      = v10;
            Cb[(long long)(rB + r + 16) * ldc + cB + 16] = v11;
        }
    }
}

// ---------------------------------------------------------------------------
// Embedding lookup + sinusoidal positional encoding
// ---------------------------------------------------------------------------
__global__ __launch_bounds__(256)
void embed_pe_kernel(const int* __restrict__ x, const float* __restrict__ emb,
                     float* __restrict__ h, int total)
{
    int idx = blockIdx.x * 256 + threadIdx.x;
    if (idx >= total) return;
    int d  = idx & (DMODEL - 1);
    int bs = idx >> 10;                 // log2(DMODEL)
    int s  = bs & (SEQ - 1);
    int tok = x[bs];
    float pos = (float)s;
    float two_i = (float)((d >> 1) << 1);               // 2*(d/2)
    float angle = pos * __expf(-(two_i / (float)DMODEL) * 9.210340371976184f); // ln(10000)
    float pe = (d & 1) ? __cosf(angle) : __sinf(angle);
    h[idx] = emb[(long long)tok * DMODEL + d] + pe;
}

// ---------------------------------------------------------------------------
// Fold W_O: concat of 16 identical heads @ W_O  ==  head @ Wred,
// Wred[h64, e] = sum_t W_O[t*64 + h64, e]
// ---------------------------------------------------------------------------
__global__ __launch_bounds__(256)
void wo_reduce_kernel(const float* __restrict__ W_O, float* __restrict__ Wred)
{
    int idx = blockIdx.x * 256 + threadIdx.x;
    if (idx >= HDIM * DMODEL) return;
    int h64 = idx >> 10;
    int e   = idx & (DMODEL - 1);
    float s = 0.0f;
    #pragma unroll
    for (int t = 0; t < NHEAD; ++t)
        s += W_O[(long long)(t * HDIM + h64) * DMODEL + e];
    Wred[idx] = s;
}

// ---------------------------------------------------------------------------
// In-place row softmax (one workgroup per row)
// ---------------------------------------------------------------------------
__global__ __launch_bounds__(256)
void softmax_rows_kernel(float* __restrict__ x, int cols)
{
    __shared__ float red[256];
    float* p = x + (long long)blockIdx.x * cols;
    int tid = threadIdx.x;

    float m = -3.402823466e38f;
    for (int c = tid; c < cols; c += 256) m = fmaxf(m, p[c]);
    red[tid] = m; __syncthreads();
    for (int s = 128; s > 0; s >>= 1) {
        if (tid < s) red[tid] = fmaxf(red[tid], red[tid + s]);
        __syncthreads();
    }
    m = red[0]; __syncthreads();

    float sum = 0.0f;
    for (int c = tid; c < cols; c += 256) {
        float e = __expf(p[c] - m);
        p[c] = e; sum += e;
    }
    red[tid] = sum; __syncthreads();
    for (int s = 128; s > 0; s >>= 1) {
        if (tid < s) red[tid] += red[tid + s];
        __syncthreads();
    }
    float inv = 1.0f / red[0];
    for (int c = tid; c < cols; c += 256) p[c] *= inv;
}

// ---------------------------------------------------------------------------
// out = LayerNorm(X + R) * g + b  (D = 1024, one workgroup per row)
// ---------------------------------------------------------------------------
__global__ __launch_bounds__(256)
void add_layernorm_kernel(const float* __restrict__ X, const float* __restrict__ R,
                          const float* __restrict__ g, const float* __restrict__ b,
                          float* __restrict__ out)
{
    __shared__ float red[256];
    long long row = blockIdx.x;
    const float* xr = X + row * DMODEL;
    const float* rr = R + row * DMODEL;
    int tid = threadIdx.x;

    float v[4];
    float s = 0.0f;
    #pragma unroll
    for (int i = 0; i < 4; ++i) {
        v[i] = xr[tid + i * 256] + rr[tid + i * 256];
        s += v[i];
    }
    red[tid] = s; __syncthreads();
    for (int st = 128; st > 0; st >>= 1) {
        if (tid < st) red[tid] += red[tid + st];
        __syncthreads();
    }
    float mu = red[0] * (1.0f / DMODEL); __syncthreads();

    float s2 = 0.0f;
    #pragma unroll
    for (int i = 0; i < 4; ++i) { float d = v[i] - mu; s2 += d * d; }
    red[tid] = s2; __syncthreads();
    for (int st = 128; st > 0; st >>= 1) {
        if (tid < st) red[tid] += red[tid + st];
        __syncthreads();
    }
    float rstd = rsqrtf(red[0] * (1.0f / DMODEL) + 1e-5f);

    #pragma unroll
    for (int i = 0; i < 4; ++i) {
        int c = tid + i * 256;
        out[row * DMODEL + c] = (v[i] - mu) * rstd * g[c] + b[c];
    }
}

// ---------------------------------------------------------------------------
// In-place row log-softmax (one workgroup per row of VOCAB cols)
// ---------------------------------------------------------------------------
__global__ __launch_bounds__(256)
void log_softmax_rows_kernel(float* __restrict__ x, int cols)
{
    __shared__ float red[256];
    float* p = x + (long long)blockIdx.x * cols;
    int tid = threadIdx.x;

    float m = -3.402823466e38f;
    for (int c = tid; c < cols; c += 256) m = fmaxf(m, p[c]);
    red[tid] = m; __syncthreads();
    for (int s = 128; s > 0; s >>= 1) {
        if (tid < s) red[tid] = fmaxf(red[tid], red[tid + s]);
        __syncthreads();
    }
    m = red[0]; __syncthreads();

    float sum = 0.0f;
    for (int c = tid; c < cols; c += 256) sum += __expf(p[c] - m);
    red[tid] = sum; __syncthreads();
    for (int s = 128; s > 0; s >>= 1) {
        if (tid < s) red[tid] += red[tid + s];
        __syncthreads();
    }
    float lse = m + __logf(red[0]);
    for (int c = tid; c < cols; c += 256) p[c] = p[c] - lse;
}

// ---------------------------------------------------------------------------
// Orchestration
// ---------------------------------------------------------------------------
extern "C" void kernel_launch(void* const* d_in, const int* in_sizes, int n_in,
                              void* d_out, int out_size, void* d_ws, size_t ws_size,
                              hipStream_t stream)
{
    const int*   x     = (const int*)  d_in[0];
    const float* emb   = (const float*)d_in[1];
    const float* W_Q   = (const float*)d_in[2];
    const float* W_K   = (const float*)d_in[3];
    const float* W_V   = (const float*)d_in[4];
    const float* W_O   = (const float*)d_in[5];
    const float* ln_g  = (const float*)d_in[6];
    const float* ln_b  = (const float*)d_in[7];
    const float* w1    = (const float*)d_in[8];
    const float* b1    = (const float*)d_in[9];
    const float* w2    = (const float*)d_in[10];
    const float* b2    = (const float*)d_in[11];
    const float* out_w = (const float*)d_in[12];
    const float* out_b = (const float*)d_in[13];
    float* logits = (float*)d_out;

    const int BS = BATCH * SEQ;  // 4096 rows

    // workspace layout (float offsets)
    float* ws   = (float*)d_ws;
    float* h    = ws;                               // 4096*1024
    float* h2   = h    + (long long)BS * DMODEL;    // 4096*1024
    float* q    = h2   + (long long)BS * DMODEL;    // 4096*64
    float* kbuf = q    + (long long)BS * HDIM;
    float* vbuf = kbuf + (long long)BS * HDIM;
    float* head = vbuf + (long long)BS * HDIM;      // 4096*64
    float* wred = head + (long long)BS * HDIM;      // 64*1024
    float* sc   = wred + (long long)HDIM * DMODEL;  // 2*2048*2048 (reused: a, ff1, h3)

    auto gemm = [&](const float* A, const float* B, float* C, const float* bias,
                    int M, int N, int K, int lda, int ldb, int ldc,
                    int transB, float alpha, int relu,
                    long long sA, long long sB, long long sC, int batch) {
        dim3 grid(N / TN, M / TM, batch);
        if (transB)
            gemm_f16wmma<1><<<grid, 256, 0, stream>>>(A, B, C, bias, M, N, K,
                                                      lda, ldb, ldc, alpha, relu, sA, sB, sC);
        else
            gemm_f16wmma<0><<<grid, 256, 0, stream>>>(A, B, C, bias, M, N, K,
                                                      lda, ldb, ldc, alpha, relu, sA, sB, sC);
    };

    // 1. h = emb[x] + PE
    {
        int total = BS * DMODEL;
        embed_pe_kernel<<<(total + 255) / 256, 256, 0, stream>>>(x, emb, h, total);
    }
    // 2. q/k/v = h @ W_{Q,K,V}   (M=4096, N=64, K=1024)
    gemm(h, W_Q, q,    nullptr, BS, HDIM, DMODEL, DMODEL, HDIM, HDIM, 0, 1.0f, 0, 0, 0, 0, 1);
    gemm(h, W_K, kbuf, nullptr, BS, HDIM, DMODEL, DMODEL, HDIM, HDIM, 0, 1.0f, 0, 0, 0, 0, 1);
    gemm(h, W_V, vbuf, nullptr, BS, HDIM, DMODEL, DMODEL, HDIM, HDIM, 0, 1.0f, 0, 0, 0, 0, 1);
    // 3. scores = q @ k^T / 8   (batched over B, M=N=2048, K=64)
    gemm(q, kbuf, sc, nullptr, SEQ, SEQ, HDIM, HDIM, HDIM, SEQ, 1, 0.125f, 0,
         (long long)SEQ * HDIM, (long long)SEQ * HDIM, (long long)SEQ * SEQ, BATCH);
    // 4. softmax rows (4096 rows of 2048)
    softmax_rows_kernel<<<BS, 256, 0, stream>>>(sc, SEQ);
    // 5. head = attn @ v   (batched, M=2048, N=64, K=2048)
    gemm(sc, vbuf, head, nullptr, SEQ, HDIM, SEQ, SEQ, HDIM, HDIM, 0, 1.0f, 0,
         (long long)SEQ * SEQ, (long long)SEQ * HDIM, (long long)SEQ * HDIM, BATCH);
    // 6. fold W_O (16 identical heads concatenated -> head @ Wred)
    wo_reduce_kernel<<<(HDIM * DMODEL + 255) / 256, 256, 0, stream>>>(W_O, wred);
    // 7. a = head @ Wred   (M=4096, N=1024, K=64) -> reuse sc region
    float* abuf = sc;
    gemm(head, wred, abuf, nullptr, BS, DMODEL, HDIM, HDIM, DMODEL, DMODEL, 0, 1.0f, 0, 0, 0, 0, 1);
    // 8. h2 = LN(a + h)
    add_layernorm_kernel<<<BS, 256, 0, stream>>>(abuf, h, ln_g, ln_b, h2);
    // 9. ff1 = relu(h2 @ w1 + b1)   (M=4096, N=2048, K=1024) -> sc region
    float* ff1 = sc;
    gemm(h2, w1, ff1, b1, BS, FFDIM, DMODEL, DMODEL, FFDIM, FFDIM, 0, 1.0f, 1, 0, 0, 0, 1);
    // 10. ff2 = ff1 @ w2 + b2   (M=4096, N=1024, K=2048) -> h region (free now)
    float* ff2 = h;
    gemm(ff1, w2, ff2, b2, BS, DMODEL, FFDIM, FFDIM, DMODEL, DMODEL, 0, 1.0f, 0, 0, 0, 0, 1);
    // 11. h3 = LN(ff2 + h2) -> sc region (free after step 10 consumed ff1)
    float* h3 = sc;
    add_layernorm_kernel<<<BS, 256, 0, stream>>>(ff2, h2, ln_g, ln_b, h3);
    // 12. logits = h3 @ out_w + out_b   (M=4096, N=32000, K=1024) -> d_out
    gemm(h3, out_w, logits, out_b, BS, VOCAB, DMODEL, DMODEL, VOCAB, VOCAB, 0, 1.0f, 0, 0, 0, 0, 1);
    // 13. log_softmax rows in place (4096 rows of 32000)
    log_softmax_rows_kernel<<<BS, 256, 0, stream>>>(logits, VOCAB);

    (void)in_sizes; (void)n_in; (void)out_size; (void)ws_size;
}
